// TextRNN_28939489640951
// MI455X (gfx1250) — matrix operand order repaired and checked by reading, not compiled, not run on hardware
//
#include <hip/hip_runtime.h>
#include <hip/hip_bf16.h>
#include <math.h>

#define VOCAB  32000
#define EMB    256
#define HID    512
#define MAXLEN 512
#define BATCH  128
#define NCLASS 2
#define NPERS  8   // K-tiles of W_hh held persistently in VGPRs (of 16 total)

typedef __attribute__((ext_vector_type(16))) __bf16 bf16x16;
typedef __attribute__((ext_vector_type(8)))  __bf16 bf16x8;
typedef __attribute__((ext_vector_type(8)))  float  f32x8;
typedef __attribute__((ext_vector_type(4)))  float  f32x4;

// B fragment (kt, j) relative to a per-wave/lane base pointer:
// stride per K-tile = TN(=32) tiles * 32 lanes * 16 bf16 = 16384 elements.
#define LOADB(P, kt, j) \
  (*(const bf16x16*)((P) + (size_t)(kt) * 16384 + (j) * 512))

// CDNA5 hardware tanh (TRANS op). Trailing v_nop satisfies the ISA's
// "1 independent op after a TRANS before its result is consumed" rule,
// which the compiler cannot enforce through inline asm.
__device__ __forceinline__ float fast_tanh(float x) {
  float r;
  asm volatile("v_tanh_f32 %0, %1\n\tv_nop" : "=v"(r) : "v"(x));
  return r;
}

// ---------------------------------------------------------------------------
// Weight repack: row-major f32 W[K][N] -> WMMA-B-fragment-contiguous bf16.
// Tile (tk,tn) covers K rows [tk*32,+32), N cols [tn*16,+16).
// Lane l of a wave owns 16 contiguous bf16: W[tk*32 + (l>>4)*8 + (i<8 ? i : i+8)][tn*16 + (l&15)]
// ---------------------------------------------------------------------------
__global__ void pack_B_kernel(const float* __restrict__ W, __bf16* __restrict__ out,
                              int K, int N) {
  int tid  = blockIdx.x * blockDim.x + threadIdx.x;
  int TN   = N >> 4;
  int ntil = (K >> 5) * TN;
  int tile = tid >> 5;
  int lane = tid & 31;
  if (tile >= ntil) return;
  int tk = tile / TN, tn = tile % TN;
  int n     = tn * 16 + (lane & 15);
  int kbase = tk * 32 + (lane >> 4) * 8;
  __bf16* dst = out + ((size_t)tile * 32 + lane) * 16;
#pragma unroll
  for (int i = 0; i < 16; ++i)
    dst[i] = (__bf16)W[(size_t)(kbase + (i < 8 ? i : i + 8)) * N + n];
}

// A-fragment (16x32 bf16) from a row-major f32 row, converting on the fly.
// Lane l holds row m=l&15; kh=l>>4: lo8 at K=k0+kh*8, hi8 at K=k0+16+kh*8.
__device__ __forceinline__ bf16x16 load_a_frag_f32(const float* __restrict__ row,
                                                   int k0, int kh) {
  const float* p0 = row + k0 + kh * 8;
  const float* p1 = row + k0 + 16 + kh * 8;
  f32x4 v0 = *(const f32x4*)(p0);
  f32x4 v1 = *(const f32x4*)(p0 + 4);
  f32x4 v2 = *(const f32x4*)(p1);
  f32x4 v3 = *(const f32x4*)(p1 + 4);
  bf16x16 a;
#pragma unroll
  for (int i = 0; i < 4; ++i) {
    a[i]      = (__bf16)v0[i];
    a[4 + i]  = (__bf16)v1[i];
    a[8 + i]  = (__bf16)v2[i];
    a[12 + i] = (__bf16)v3[i];
  }
  return a;
}

// A-fragment from a bf16 LDS row (16B-aligned chunks -> ds_load_b128).
__device__ __forceinline__ bf16x16 load_a_frag_lds(const __bf16* row, int k0, int kh) {
  bf16x8 lo = *(const bf16x8*)(row + k0 + kh * 8);
  bf16x8 hi = *(const bf16x8*)(row + k0 + 16 + kh * 8);
  bf16x16 a;
#pragma unroll
  for (int i = 0; i < 8; ++i) { a[i] = lo[i]; a[8 + i] = hi[i]; }
  return a;
}

// ---------------------------------------------------------------------------
// Phase 1 (parallel, full chip): pre[t][b][h] = emb[ids[b,t]] @ W_ih
// Grid: MAXLEN * (BATCH/16) = 4096 workgroups, 256 threads, wave owns 4 N-tiles.
// ---------------------------------------------------------------------------
__global__ __launch_bounds__(256)
void precompute_xW_kernel(const int* __restrict__ ids,
                          const float* __restrict__ emb,
                          const __bf16* __restrict__ WihP,
                          float* __restrict__ pre) {
  const int t    = blockIdx.x >> 3;
  const int wgb  = blockIdx.x & 7;
  const int wave = threadIdx.x >> 5;
  const int lane = threadIdx.x & 31;
  const int m    = lane & 15;
  const int kh   = lane >> 4;

  const int id = ids[(wgb * 16 + m) * MAXLEN + t];
  const float* xrow  = emb + (size_t)id * EMB;
  const __bf16* bbase = WihP + ((size_t)(wave * 4) * 32 + lane) * 16;

  f32x8 acc[4] = {};
  bf16x16 bcur[4], bnxt[4];
#pragma unroll
  for (int j = 0; j < 4; ++j) bcur[j] = LOADB(bbase, 0, j);
#pragma unroll
  for (int kt = 0; kt < EMB / 32; ++kt) {
    if (kt + 1 < EMB / 32) {
#pragma unroll
      for (int j = 0; j < 4; ++j) bnxt[j] = LOADB(bbase, kt + 1, j);
    }
    bf16x16 a = load_a_frag_f32(xrow, kt * 32, kh);
#pragma unroll
    for (int j = 0; j < 4; ++j)
      acc[j] = __builtin_amdgcn_wmma_f32_16x16x32_bf16(
          false, a, false, bcur[j], (short)0, acc[j], false, false);
#pragma unroll
    for (int j = 0; j < 4; ++j) bcur[j] = bnxt[j];
  }
  // D layout: element r of lane l -> row (r + 8*(l>>4)), col (l&15)
  const int mr = kh * 8, col = lane & 15;
#pragma unroll
  for (int j = 0; j < 4; ++j) {
    int n0 = (wave * 4 + j) * 16;
    float* prow = pre + ((size_t)t * BATCH + wgb * 16 + mr) * HID + n0 + col;
#pragma unroll
    for (int r = 0; r < 8; ++r)
      prow[(size_t)r * HID] = acc[j][r];
  }
}

// ---------------------------------------------------------------------------
// Phase 2 (sequential scan). Persistent-RNN style: each workgroup owns 16
// batch rows; h lives in LDS (bf16, double buffered); HALF of the packed
// W_hh B-fragments (K-tiles 0..NPERS-1, 256 VGPRs/lane) are loaded ONCE into
// registers and reused for all 512 steps; the other half is streamed from L2
// with its loads issued at step start so they overlap the register-resident
// WMMA section. h_{t+1} = tanh(pre_t + h_t @ W_hh + b_h).
// ---------------------------------------------------------------------------
__global__ __launch_bounds__(256)
void rnn_scan_kernel(const int* __restrict__ ids,
                     const float* __restrict__ emb,
                     const __bf16* __restrict__ WihP,
                     const __bf16* __restrict__ WhhP,
                     const float* __restrict__ b_h,
                     const float* __restrict__ W_out,
                     const float* __restrict__ b_out,
                     const float* __restrict__ pre,   // may be nullptr
                     float* __restrict__ out) {
  __shared__ __bf16 hbuf[2][16][HID];          // 32 KB double-buffered hidden state
  __shared__ float  partial[16][NCLASS][8];    // classifier reduction scratch

  const int wgb  = blockIdx.x;          // batch chunk (16 rows)
  const int wave = threadIdx.x >> 5;
  const int lane = threadIdx.x & 31;
  const int m    = lane & 15;
  const int kh   = lane >> 4;
  const int mr   = kh * 8;              // D-layout row offset
  const int col  = lane & 15;

  // h_0 = 0
  for (int i = threadIdx.x; i < 16 * HID; i += blockDim.x)
    ((__bf16*)hbuf[0])[i] = (__bf16)0.0f;

  float bias[4];
#pragma unroll
  for (int j = 0; j < 4; ++j) bias[j] = b_h[(wave * 4 + j) * 16 + col];

  const __bf16* bbase = WhhP + ((size_t)(wave * 4) * 32 + lane) * 16;

  // Persistent W_hh half: loaded once, live in VGPRs for the whole scan.
  bf16x16 bp[NPERS][4];
#pragma unroll
  for (int kt = 0; kt < NPERS; ++kt)
#pragma unroll
    for (int j = 0; j < 4; ++j) bp[kt][j] = LOADB(bbase, kt, j);

  __syncthreads();

  for (int t = 0; t < MAXLEN; ++t) {
    const int cur = t & 1, nxt = cur ^ 1;
    f32x8 acc[4] = {};
    f32x8 pv[4]  = {};

    if (pre) {
      // Issue the x_t @ W_ih result loads up front; they overlap the h-loop.
#pragma unroll
      for (int j = 0; j < 4; ++j) {
        int n0 = (wave * 4 + j) * 16;
        const float* prow = pre + ((size_t)t * BATCH + wgb * 16 + mr) * HID + n0 + col;
#pragma unroll
        for (int r = 0; r < 8; ++r) pv[j][r] = prow[(size_t)r * HID];
      }
    } else {
      // Fallback: fused x_t @ W_ih (embedding gather + cvt to bf16 on the fly).
      const int id = ids[(wgb * 16 + m) * MAXLEN + t];
      const float* xrow = emb + (size_t)id * EMB;
      const __bf16* xb = WihP + ((size_t)(wave * 4) * 32 + lane) * 16;
#pragma unroll 2
      for (int kt = 0; kt < EMB / 32; ++kt) {
        bf16x16 a = load_a_frag_f32(xrow, kt * 32, kh);
#pragma unroll
        for (int j = 0; j < 4; ++j) {
          bf16x16 bm = LOADB(xb, kt, j);
          acc[j] = __builtin_amdgcn_wmma_f32_16x16x32_bf16(
              false, a, false, bm, (short)0, acc[j], false, false);
        }
      }
    }

    const __bf16* hrow = hbuf[cur][m];

    // Kick off the streamed-half B loads NOW: they drain while the
    // register-resident WMMA section below executes (no memory ops there).
    bf16x16 sbuf[2][4];
#pragma unroll
    for (int j = 0; j < 4; ++j) sbuf[0][j] = LOADB(bbase, NPERS + 0, j);
#pragma unroll
    for (int j = 0; j < 4; ++j) sbuf[1][j] = LOADB(bbase, NPERS + 1, j);

    // Register-resident section: K-tiles 0..NPERS-1, zero memory traffic.
#pragma unroll
    for (int kt = 0; kt < NPERS; ++kt) {
      bf16x16 a = load_a_frag_lds(hrow, kt * 32, kh);
#pragma unroll
      for (int j = 0; j < 4; ++j)
        acc[j] = __builtin_amdgcn_wmma_f32_16x16x32_bf16(
            false, a, false, bp[kt][j], (short)0, acc[j], false, false);
    }

    // Streamed section: K-tiles NPERS..15, 2-deep software pipeline from L2.
#pragma unroll
    for (int kt = NPERS; kt < HID / 32; ++kt) {
      bf16x16 a = load_a_frag_lds(hrow, kt * 32, kh);
      bf16x16 c4[4];
#pragma unroll
      for (int j = 0; j < 4; ++j) c4[j] = sbuf[(kt - NPERS) & 1][j];
      if (kt + 2 < HID / 32) {
#pragma unroll
        for (int j = 0; j < 4; ++j) sbuf[(kt - NPERS) & 1][j] = LOADB(bbase, kt + 2, j);
      }
#pragma unroll
      for (int j = 0; j < 4; ++j)
        acc[j] = __builtin_amdgcn_wmma_f32_16x16x32_bf16(
            false, a, false, c4[j], (short)0, acc[j], false, false);
    }

    // Epilogue: h_{t+1} = tanh(acc + pre + bias) -> bf16 into the other buffer.
#pragma unroll
    for (int j = 0; j < 4; ++j) {
      int n0 = (wave * 4 + j) * 16;
#pragma unroll
      for (int r = 0; r < 8; ++r) {
        float v = fast_tanh(acc[j][r] + pv[j][r] + bias[j]);
        hbuf[nxt][mr + r][n0 + col] = (__bf16)v;
      }
    }
    __syncthreads();
  }

  // Classifier head, 8-way K-split over all 256 threads + LDS reduction.
  // MAXLEN is even, so the final state lives in buffer 0.
  {
    int b  = threadIdx.x >> 4;         // 0..15
    int c  = (threadIdx.x >> 3) & 1;   // 0..1
    int ch = threadIdx.x & 7;          // 0..7 -> K chunk of 64
    const __bf16* hr = hbuf[MAXLEN & 1][b] + ch * 64;
    const float*  wo = W_out + (size_t)ch * 64 * NCLASS + c;
    float s = 0.0f;
#pragma unroll 4
    for (int k = 0; k < 64; ++k)
      s += (float)hr[k] * wo[k * NCLASS];
    partial[b][c][ch] = s;
  }
  __syncthreads();
  if (threadIdx.x < 32) {
    int b = threadIdx.x >> 1;
    int c = threadIdx.x & 1;
    float s = b_out[c];
#pragma unroll
    for (int ch = 0; ch < 8; ++ch) s += partial[b][c][ch];
    out[(wgb * 16 + b) * NCLASS + c] = 1.0f / (1.0f + expf(-s));
  }
}

// ---------------------------------------------------------------------------
extern "C" void kernel_launch(void* const* d_in, const int* in_sizes, int n_in,
                              void* d_out, int out_size, void* d_ws, size_t ws_size,
                              hipStream_t stream) {
  const int*   ids   = (const int*)  d_in[0];
  const float* emb   = (const float*)d_in[1];
  const float* W_ih  = (const float*)d_in[2];
  const float* W_hh  = (const float*)d_in[3];
  const float* b_h   = (const float*)d_in[4];
  const float* W_out = (const float*)d_in[5];
  const float* b_out = (const float*)d_in[6];
  float* out = (float*)d_out;

  char* ws = (char*)d_ws;
  __bf16* WihP = (__bf16*)(ws);                 // 256 KB packed bf16 W_ih
  __bf16* WhhP = (__bf16*)(ws + (256 << 10));   // 512 KB packed bf16 W_hh
  float*  pre  = (float*)(ws + (1 << 20));      // 128 MB pre-activations
  const size_t pre_bytes = (size_t)MAXLEN * BATCH * HID * sizeof(float);
  const bool use_pre = ws_size >= ((size_t)(1 << 20) + pre_bytes);

  // One-time-cost weight repack (re-run every call for determinism).
  const int tiles_ih = (EMB / 32) * (HID / 16);  // 256 tiles -> 8192 threads
  const int tiles_hh = (HID / 32) * (HID / 16);  // 512 tiles -> 16384 threads
  pack_B_kernel<<<(tiles_ih * 32 + 255) / 256, 256, 0, stream>>>(W_ih, WihP, EMB, HID);
  pack_B_kernel<<<(tiles_hh * 32 + 255) / 256, 256, 0, stream>>>(W_hh, WhhP, HID, HID);

  if (use_pre) {
    // Full-chip parallel phase: 4096 workgroups.
    precompute_xW_kernel<<<MAXLEN * (BATCH / 16), 256, 0, stream>>>(ids, emb, WihP, pre);
  }

  // Sequential scan: 8 workgroups (one per 16 batch rows), 8 waves each.
  rnn_scan_kernel<<<BATCH / 16, 256, 0, stream>>>(
      ids, emb, WihP, WhhP, b_h, W_out, b_out, use_pre ? pre : nullptr, out);
}
